// SequentialMambaEncoder_17575006175958
// MI455X (gfx1250) — matrix operand order repaired and checked by reading, not compiled, not run on hardware
//
#include <hip/hip_runtime.h>

// ---------------------------------------------------------------------------
// Mamba encoder, MI455X (gfx1250, wave32, WMMA).
// Exact-algorithm optimization: the 256 context positions are shared across
// all 64 targets per batch element (causal scan/conv), so we prefill the
// context once per layer, save SSM state + conv tail, then decode the 512
// targets as single steps. ~9 GFLOP total, fits in L2 -> latency bound.
// GEMMs: v_wmma_f32_16x16x32_bf16; A-frags = 2x ds_load_b128 from row-major
// LDS tiles; B-frags = 2x global_load_b128 from weights pre-repacked into
// fragment order (done once in cvt_weights).
// ---------------------------------------------------------------------------

#define NLAY 4
#define DMODEL 256
#define DINNER 512
#define NSTATE 16
#define RRANK 16
#define KCONV 4

typedef __attribute__((ext_vector_type(16))) __bf16 bf16x16;
typedef __attribute__((ext_vector_type(8)))  float  f32x8;

struct u4x2 { uint4 a, b; };

__device__ __forceinline__ f32x8 wmma_bf16(bf16x16 a, bf16x16 b, f32x8 c) {
  return __builtin_amdgcn_wmma_f32_16x16x32_bf16(false, a, false, b, (short)0, c,
                                                 false, false);
}

// A fragment (16x32 bf16, MxK) from a row-major LDS tile.
// Lane (half=lane>>4, r=lane&15) holds k = kbase+16*half+{0..15}:
// two contiguous 16B runs -> two ds_load_b128.
__device__ __forceinline__ bf16x16 load_a_frag(const __bf16* lds, int stride, int kbase) {
  int lane = threadIdx.x & 31;
  int half = lane >> 4, r = lane & 15;
  const __bf16* p = lds + r * stride + kbase + 16 * half;
  u4x2 t;
  t.a = *(const uint4*)p;
  t.b = *(const uint4*)(p + 8);
  return __builtin_bit_cast(bf16x16, t);
}

// B fragment from fragment-order packed weights: block `blk` = (ct*nkt + kt),
// lane's 16 bf16 contiguous -> two global_load_b128 (32B/lane, coalesced).
__device__ __forceinline__ bf16x16 load_b_frag(const __bf16* wpk, int blk) {
  const __bf16* p = wpk + ((size_t)blk * 32 + (threadIdx.x & 31)) * 16;
  u4x2 t;
  t.a = *(const uint4*)p;
  t.b = *(const uint4*)(p + 8);
  return __builtin_bit_cast(bf16x16, t);
}

// --------------------------- weight preparation ----------------------------
// Repack W (K x N row-major, f32) -> bf16 fragment order:
// dst[(((ct*nkt + kt)*32 + lane)*16 + j] = W[kt*32 + 16*(lane>>4) + j][ct*16 + (lane&15)]
__global__ void cvt_weights_kernel(const float* __restrict__ W_in,
                                   const float* __restrict__ W_x,
                                   const float* __restrict__ W_dt,
                                   const float* __restrict__ W_out,
                                   const float* __restrict__ A_log,
                                   __bf16* __restrict__ Wi_bf, __bf16* __restrict__ Wx_bf,
                                   __bf16* __restrict__ Wdt_bf, __bf16* __restrict__ Wo_bf,
                                   float* __restrict__ Aneg) {
  int stride = gridDim.x * blockDim.x;
  int gid = blockIdx.x * blockDim.x + threadIdx.x;

  // W_in: K=256 (nkt=8), N=1024 (nct=64)
  for (int i = gid; i < NLAY * 256 * 1024; i += stride) {
    int l = i >> 18, rem = i & 262143;
    int j = rem & 15, lane = (rem >> 4) & 31, blk = rem >> 9;
    int kt = blk & 7, ct = blk >> 3;
    int k = kt * 32 + 16 * (lane >> 4) + j, c = ct * 16 + (lane & 15);
    Wi_bf[i] = (__bf16)W_in[(l * 256 + k) * 1024 + c];
  }
  // W_x: K=512 (nkt=16), N=48 (nct=3)
  for (int i = gid; i < NLAY * 512 * 48; i += stride) {
    int l = i / 24576, rem = i % 24576;
    int j = rem & 15, lane = (rem >> 4) & 31, blk = rem >> 9;
    int kt = blk & 15, ct = blk >> 4;
    int k = kt * 32 + 16 * (lane >> 4) + j, c = ct * 16 + (lane & 15);
    Wx_bf[i] = (__bf16)W_x[(l * 512 + k) * 48 + c];
  }
  // W_dt: K=16 padded to 32 (nkt=1), N=512 (nct=32)
  for (int i = gid; i < NLAY * 32 * 512; i += stride) {
    int l = i >> 14, rem = i & 16383;
    int j = rem & 15, lane = (rem >> 4) & 31, ct = rem >> 9;
    int k = 16 * (lane >> 4) + j, c = ct * 16 + (lane & 15);
    Wdt_bf[i] = (k < RRANK) ? (__bf16)W_dt[(l * RRANK + k) * 512 + c] : (__bf16)0.0f;
  }
  // W_out: K=512 (nkt=16), N=256 (nct=16)
  for (int i = gid; i < NLAY * 512 * 256; i += stride) {
    int l = i >> 17, rem = i & 131071;
    int j = rem & 15, lane = (rem >> 4) & 31, blk = rem >> 9;
    int kt = blk & 15, ct = blk >> 4;
    int k = kt * 32 + 16 * (lane >> 4) + j, c = ct * 16 + (lane & 15);
    Wo_bf[i] = (__bf16)W_out[(l * 512 + k) * 256 + c];
  }
  for (int i = gid; i < NLAY * DINNER * NSTATE; i += stride)
    Aneg[i] = -__expf(A_log[i]);
}

__global__ void copy_f32_kernel(const float* __restrict__ src, float* __restrict__ dst, int n) {
  int stride = gridDim.x * blockDim.x;
  for (int i = blockIdx.x * blockDim.x + threadIdx.x; i < n; i += stride) dst[i] = src[i];
}

// --------------------- LayerNorm + x @ W_in (256 -> 1024) ------------------
// block: 256 thr (8 waves). grid: (rows/16, 8). wave w + group cg -> col tile
__global__ void __launch_bounds__(256) ln_gemm_in_kernel(
    const float* __restrict__ X, const float* __restrict__ gamma,
    const float* __restrict__ beta, const __bf16* __restrict__ Wb,
    float* __restrict__ XZ) {
  __shared__ float ldsX[16 * 256];
  __shared__ __attribute__((aligned(16))) __bf16 ldsA[16 * 256];
  __shared__ float mu[16], rs[16];
  int tid = threadIdx.x;
  int row0 = blockIdx.x * 16;

  for (int i = tid; i < 16 * 256; i += 256)
    ldsX[i] = X[(row0 + (i >> 8)) * 256 + (i & 255)];
  __syncthreads();
  if (tid < 16) {
    const float* rp = &ldsX[tid * 256];
    float s = 0.f, s2 = 0.f;
    for (int c = 0; c < 256; ++c) { float v = rp[c]; s += v; s2 += v * v; }
    float m = s * (1.0f / 256.0f);
    float var = s2 * (1.0f / 256.0f) - m * m;
    mu[tid] = m;
    rs[tid] = rsqrtf(var + 1e-5f);
  }
  __syncthreads();
  for (int i = tid; i < 16 * 256; i += 256) {
    int r = i >> 8, c = i & 255;
    ldsA[i] = (__bf16)((ldsX[i] - mu[r]) * rs[r] * gamma[c] + beta[c]);
  }
  __syncthreads();

  int ct = blockIdx.y * 8 + (tid >> 5);  // 64 col tiles
  f32x8 acc = {};
#pragma unroll
  for (int kk = 0; kk < 8; ++kk) {
    bf16x16 a = load_a_frag(ldsA, 256, kk * 32);
    bf16x16 b = load_b_frag(Wb, ct * 8 + kk);
    acc = wmma_bf16(a, b, acc);
  }
  int lane = tid & 31, half = lane >> 4, n = lane & 15;
  int col = ct * 16 + n;
#pragma unroll
  for (int i = 0; i < 8; ++i)
    XZ[(row0 + i + 8 * half) * 1024 + col] = acc[i];
}

// ------------------- causal depthwise conv (K=4) + SiLU --------------------
__global__ void conv_ctx_kernel(const float* __restrict__ XZ, const float* __restrict__ Wc,
                                const float* __restrict__ bc, float* __restrict__ XCV,
                                float* __restrict__ tail, int Lseq) {
  int idx = blockIdx.x * 256 + threadIdx.x;  // rows * 512
  int row = idx >> 9, d = idx & 511;
  int s = row / Lseq, pos = row % Lseq;
  float acc = bc[d];
  float xcur = 0.0f;
#pragma unroll
  for (int k = 0; k < KCONV; ++k) {
    int p = pos + k - (KCONV - 1);
    float xv = (p >= 0) ? XZ[(s * Lseq + p) * 1024 + d] : 0.0f;
    acc += Wc[d * KCONV + k] * xv;
    if (k == KCONV - 1) xcur = xv;
  }
  XCV[row * 512 + d] = acc / (1.0f + __expf(-acc));  // silu
  if (pos >= Lseq - 3) tail[(s * 3 + (pos - (Lseq - 3))) * 512 + d] = xcur;
}

__global__ void conv_tgt_kernel(const float* __restrict__ XZ, const float* __restrict__ Wc,
                                const float* __restrict__ bc, const float* __restrict__ tail,
                                float* __restrict__ XCV) {
  int idx = blockIdx.x * 256 + threadIdx.x;  // 512 * 512
  int t = idx >> 9, d = idx & 511;
  int b = t >> 6;
  float acc = bc[d] + Wc[d * 4 + 0] * tail[(b * 3 + 0) * 512 + d] +
              Wc[d * 4 + 1] * tail[(b * 3 + 1) * 512 + d] +
              Wc[d * 4 + 2] * tail[(b * 3 + 2) * 512 + d] +
              Wc[d * 4 + 3] * XZ[t * 1024 + d];
  XCV[t * 512 + d] = acc / (1.0f + __expf(-acc));
}

// ------------------------ xcv @ W_x (512 -> 48) ----------------------------
// block: 96 thr (3 waves, one per 16-col tile). grid: rows/16
__global__ void __launch_bounds__(96) gemm_dbl_kernel(const float* __restrict__ XCV,
                                                      const __bf16* __restrict__ Wx,
                                                      float* __restrict__ DBL) {
  __shared__ __attribute__((aligned(16))) __bf16 ldsA[16 * 512];
  int tid = threadIdx.x;
  int row0 = blockIdx.x * 16;
  for (int i = tid; i < 16 * 512; i += 96)
    ldsA[i] = (__bf16)XCV[(row0 + (i >> 9)) * 512 + (i & 511)];
  __syncthreads();
  int ct = tid >> 5;  // 3 col tiles
  f32x8 acc = {};
#pragma unroll
  for (int kk = 0; kk < 16; ++kk) {
    bf16x16 a = load_a_frag(ldsA, 512, kk * 32);
    bf16x16 b = load_b_frag(Wx, ct * 16 + kk);
    acc = wmma_bf16(a, b, acc);
  }
  int lane = tid & 31, half = lane >> 4, n = lane & 15;
#pragma unroll
  for (int i = 0; i < 8; ++i)
    DBL[(row0 + i + 8 * half) * 48 + ct * 16 + n] = acc[i];
}

// -------------- dt = softplus(dbl[:, :16] @ W_dt + b_dt) -------------------
// K padded to 32 (single WMMA). block: 256 thr. grid: (rows/16, 4)
__global__ void __launch_bounds__(256) gemm_dt_kernel(const float* __restrict__ DBL,
                                                      const __bf16* __restrict__ Wdt,
                                                      const float* __restrict__ bdt,
                                                      float* __restrict__ DT) {
  __shared__ __attribute__((aligned(16))) __bf16 ldsA[16 * 32];
  int tid = threadIdx.x;
  int row0 = blockIdx.x * 16;
  for (int i = tid; i < 512; i += 256) {
    int r = i >> 5, c = i & 31;
    ldsA[i] = (c < RRANK) ? (__bf16)DBL[(row0 + r) * 48 + c] : (__bf16)0.0f;
  }
  __syncthreads();
  int ct = blockIdx.y * 8 + (tid >> 5);  // 32 col tiles
  f32x8 acc = {};
  bf16x16 a = load_a_frag(ldsA, 32, 0);
  bf16x16 b = load_b_frag(Wdt, ct);
  acc = wmma_bf16(a, b, acc);
  int lane = tid & 31, half = lane >> 4, n = lane & 15;
  int col = ct * 16 + n;
#pragma unroll
  for (int i = 0; i < 8; ++i) {
    float v = acc[i] + bdt[col];
    DT[(row0 + i + 8 * half) * 512 + col] =
        fmaxf(v, 0.0f) + log1pf(__expf(-fabsf(v)));  // softplus
  }
}

// --------------------------- selective-scan --------------------------------
// grid (nseq, 2): each block owns 256 channels of one sequence; 1 ch/thread,
// 16 states in registers. B/C broadcast per step via LDS. In-place: xcv -> y.
__global__ void __launch_bounds__(256) scan_kernel(
    float* __restrict__ XCV, const float* __restrict__ DBL,
    const float* __restrict__ DT, const float* __restrict__ XZ,
    const float* __restrict__ Aneg, const float* __restrict__ Dp,
    float* __restrict__ h_save, const float* __restrict__ h_init,
    int L, int rows_per_state) {
  __shared__ float BC[32];
  int seq = blockIdx.x;
  int d = blockIdx.y * 256 + threadIdx.x;
  int rowbase = seq * L;
  float a[NSTATE], h[NSTATE];
#pragma unroll
  for (int n = 0; n < NSTATE; ++n) a[n] = Aneg[d * NSTATE + n];
  if (h_init) {
    int sidx = rowbase / rows_per_state;
#pragma unroll
    for (int n = 0; n < NSTATE; ++n) h[n] = h_init[(sidx * DINNER + d) * NSTATE + n];
  } else {
#pragma unroll
    for (int n = 0; n < NSTATE; ++n) h[n] = 0.0f;
  }
  float dp = Dp[d];
  for (int p = 0; p < L; ++p) {
    int row = rowbase + p;
    __syncthreads();
    if (threadIdx.x < 32) BC[threadIdx.x] = DBL[row * 48 + RRANK + threadIdx.x];
    __syncthreads();
    float dtv = DT[row * 512 + d];
    float xv = XCV[row * 512 + d];
    float dx = dtv * xv;
    float y = 0.0f;
#pragma unroll
    for (int n = 0; n < NSTATE; ++n) {
      float dA = __expf(dtv * a[n]);
      h[n] = dA * h[n] + dx * BC[n];
      y = fmaf(h[n], BC[NSTATE + n], y);
    }
    float zv = XZ[row * 1024 + 512 + d];
    float gate = zv / (1.0f + __expf(-zv));  // silu(z)
    XCV[row * 512 + d] = (y + xv * dp) * gate;
  }
  if (h_save) {
#pragma unroll
    for (int n = 0; n < NSTATE; ++n) h_save[(seq * DINNER + d) * NSTATE + n] = h[n];
  }
}

// -------------------- y @ W_out (512 -> 256) + residual --------------------
// block: 256 thr (8 waves). grid: (rows/16, 2)
__global__ void __launch_bounds__(256) gemm_out_kernel(const float* __restrict__ Y,
                                                       const __bf16* __restrict__ Wo,
                                                       float* __restrict__ X) {
  __shared__ __attribute__((aligned(16))) __bf16 ldsA[16 * 512];
  int tid = threadIdx.x;
  int row0 = blockIdx.x * 16;
  for (int i = tid; i < 16 * 512; i += 256)
    ldsA[i] = (__bf16)Y[(row0 + (i >> 9)) * 512 + (i & 511)];
  __syncthreads();
  int ct = blockIdx.y * 8 + (tid >> 5);  // 16 col tiles
  f32x8 acc = {};
#pragma unroll
  for (int kk = 0; kk < 16; ++kk) {
    bf16x16 a = load_a_frag(ldsA, 512, kk * 32);
    bf16x16 b = load_b_frag(Wo, ct * 16 + kk);
    acc = wmma_bf16(a, b, acc);
  }
  int lane = tid & 31, half = lane >> 4, n = lane & 15;
  int col = ct * 16 + n;
#pragma unroll
  for (int i = 0; i < 8; ++i) {
    int r = row0 + i + 8 * half;
    X[r * 256 + col] += acc[i];  // residual
  }
}

// ------------------------------- host side ---------------------------------
extern "C" void kernel_launch(void* const* d_in, const int* in_sizes, int n_in,
                              void* d_out, int out_size, void* d_ws, size_t ws_size,
                              hipStream_t stream) {
  const float* xc     = (const float*)d_in[0];   // (8,256,256)
  const float* xt     = (const float*)d_in[1];   // (8,64,256)
  const float* ln_g   = (const float*)d_in[2];   // (4,256)
  const float* ln_b   = (const float*)d_in[3];   // (4,256)
  const float* W_in   = (const float*)d_in[4];   // (4,256,1024)
  const float* W_conv = (const float*)d_in[5];   // (4,512,1,4)
  const float* b_conv = (const float*)d_in[6];   // (4,512)
  const float* W_x    = (const float*)d_in[7];   // (4,512,48)
  const float* W_dt   = (const float*)d_in[8];   // (4,16,512)
  const float* b_dt   = (const float*)d_in[9];   // (4,512)
  const float* A_log  = (const float*)d_in[10];  // (4,512,16)
  const float* Dp     = (const float*)d_in[11];  // (4,512)
  const float* W_out  = (const float*)d_in[12];  // (4,512,256)

  const int RC = 8 * 256;   // context rows
  const int RT = 512;       // target rows

  char* ws = (char*)d_ws;
  size_t off = 0;
  auto alloc = [&](size_t bytes) { char* p = ws + off; off += (bytes + 255) & ~(size_t)255; return p; };

  __bf16* Wi_bf  = (__bf16*)alloc((size_t)NLAY * 256 * 1024 * 2);
  __bf16* Wx_bf  = (__bf16*)alloc((size_t)NLAY * 512 * 48 * 2);
  __bf16* Wdt_bf = (__bf16*)alloc((size_t)NLAY * 32 * 512 * 2);
  __bf16* Wo_bf  = (__bf16*)alloc((size_t)NLAY * 512 * 256 * 2);
  float* Aneg = (float*)alloc((size_t)NLAY * 512 * 16 * 4);
  float* XC   = (float*)alloc((size_t)RC * 256 * 4);
  float* XT   = (float*)alloc((size_t)RT * 256 * 4);
  float* XZc  = (float*)alloc((size_t)RC * 1024 * 4);
  float* XCVc = (float*)alloc((size_t)RC * 512 * 4);
  float* DBLc = (float*)alloc((size_t)RC * 48 * 4);
  float* DTc  = (float*)alloc((size_t)RC * 512 * 4);
  float* XZt  = (float*)alloc((size_t)RT * 1024 * 4);
  float* XCVt = (float*)alloc((size_t)RT * 512 * 4);
  float* DBLt = (float*)alloc((size_t)RT * 48 * 4);
  float* DTt  = (float*)alloc((size_t)RT * 512 * 4);
  float* TAIL = (float*)alloc((size_t)NLAY * 8 * 3 * 512 * 4);
  float* H0   = (float*)alloc((size_t)NLAY * 8 * 512 * 16 * 4);
  (void)ws_size; (void)in_sizes; (void)n_in; (void)out_size;

  cvt_weights_kernel<<<512, 256, 0, stream>>>(W_in, W_x, W_dt, W_out, A_log,
                                              Wi_bf, Wx_bf, Wdt_bf, Wo_bf, Aneg);
  copy_f32_kernel<<<512, 256, 0, stream>>>(xc, XC, RC * 256);
  copy_f32_kernel<<<512, 256, 0, stream>>>(xt, XT, RT * 256);

  for (int l = 0; l < NLAY; ++l) {
    const float* g   = ln_g + l * 256;
    const float* be  = ln_b + l * 256;
    const __bf16* Wi = Wi_bf + (size_t)l * 256 * 1024;
    const __bf16* Wx = Wx_bf + (size_t)l * 512 * 48;
    const __bf16* Wd = Wdt_bf + (size_t)l * 32 * 512;
    const __bf16* Wo = Wo_bf + (size_t)l * 512 * 256;
    const float* Wc  = W_conv + (size_t)l * 512 * 4;
    const float* bcv = b_conv + (size_t)l * 512;
    const float* bdt = b_dt + (size_t)l * 512;
    const float* An  = Aneg + (size_t)l * 512 * 16;
    const float* dpv = Dp + (size_t)l * 512;
    float* tail_l = TAIL + (size_t)l * 8 * 3 * 512;
    float* h0_l   = H0 + (size_t)l * 8 * 512 * 16;

    // ---- context prefill ----
    ln_gemm_in_kernel<<<dim3(RC / 16, 8), 256, 0, stream>>>(XC, g, be, Wi, XZc);
    conv_ctx_kernel<<<(RC * 512) / 256, 256, 0, stream>>>(XZc, Wc, bcv, XCVc, tail_l, 256);
    gemm_dbl_kernel<<<RC / 16, 96, 0, stream>>>(XCVc, Wx, DBLc);
    gemm_dt_kernel<<<dim3(RC / 16, 4), 256, 0, stream>>>(DBLc, Wd, bdt, DTc);
    scan_kernel<<<dim3(8, 2), 256, 0, stream>>>(XCVc, DBLc, DTc, XZc, An, dpv,
                                                h0_l, nullptr, 256, 256);
    gemm_out_kernel<<<dim3(RC / 16, 2), 256, 0, stream>>>(XCVc, Wo, XC);

    // ---- target decode (single step per token) ----
    ln_gemm_in_kernel<<<dim3(RT / 16, 8), 256, 0, stream>>>(XT, g, be, Wi, XZt);
    conv_tgt_kernel<<<(RT * 512) / 256, 256, 0, stream>>>(XZt, Wc, bcv, tail_l, XCVt);
    gemm_dbl_kernel<<<RT / 16, 96, 0, stream>>>(XCVt, Wx, DBLt);
    gemm_dt_kernel<<<dim3(RT / 16, 4), 256, 0, stream>>>(DBLt, Wd, bdt, DTt);
    scan_kernel<<<dim3(RT, 2), 256, 0, stream>>>(XCVt, DBLt, DTt, XZt, An, dpv,
                                                 nullptr, h0_l, 1, 64);
    gemm_out_kernel<<<dim3(RT / 16, 2), 256, 0, stream>>>(XCVt, Wo, XT);
  }

  copy_f32_kernel<<<512, 256, 0, stream>>>(XT, (float*)d_out, RT * 256);
}
